// MultiHeadAttention_33354716020798
// MI455X (gfx1250) — matrix operand order repaired and checked by reading, not compiled, not run on hardware
//
#include <hip/hip_runtime.h>
#include <hip/hip_bf16.h>

// ---------------------------------------------------------------------------
// Multi-head attention for MI455X (gfx1250, wave32, WMMA + async-to-LDS).
// Pipeline: f32->bf16 convert/transpose, bf16 WMMA QKV GEMMs (fp32 accum)
// with async double-buffered LDS A-tiles, flash-style streaming-softmax
// attention with async K/V LDS tiles, bf16 WMMA output projection.
// ---------------------------------------------------------------------------

typedef __bf16 bf16_t;
typedef __attribute__((ext_vector_type(16))) __bf16 v16bf;
typedef __attribute__((ext_vector_type(8)))  float  v8f;

constexpr int B_  = 4;
constexpr int S_  = 2048;
constexpr int E_  = 1024;
constexpr int H_  = 16;
constexpr int Dh_ = 64;
constexpr int BS_ = B_ * S_;     // 8192
constexpr int HD_ = H_ * Dh_;    // 1024

// ---------------------------------------------------------------------------
// Async global->LDS copy (gfx1250 GLOBAL_LOAD_ASYNC_TO_LDS_B128, ASYNCcnt).
// The builtin's pointer params are v4i32 pointers in AS1 (global) / AS3 (LDS)
// per the clang diagnostic; build those types exactly.
// ---------------------------------------------------------------------------
typedef int v4i32 __attribute__((vector_size(4 * sizeof(int))));
typedef __attribute__((address_space(1))) v4i32 as1_v4i;
typedef __attribute__((address_space(3))) v4i32 as3_v4i;

#if defined(__has_builtin)
#if __has_builtin(__builtin_amdgcn_global_load_async_to_lds_b128)
#define HAVE_ASYNC_LDS 1
#endif
#endif

__device__ __forceinline__ void async_copy_b128(const void* gsrc, void* ldst) {
#ifdef HAVE_ASYNC_LDS
  // LDS byte address = low 32 bits of the generic address (ISA aperture rule).
  __builtin_amdgcn_global_load_async_to_lds_b128(
      (as1_v4i*)(unsigned long long)gsrc,
      (as3_v4i*)(unsigned int)(unsigned long long)ldst,
      0, 0);
#else
  *(uint4*)ldst = *(const uint4*)gsrc;
#endif
}

__device__ __forceinline__ void wait_async() {
#ifdef HAVE_ASYNC_LDS
#if __has_builtin(__builtin_amdgcn_s_wait_asynccnt)
  __builtin_amdgcn_s_wait_asynccnt(0);
#else
  asm volatile("s_wait_asynccnt 0" ::: "memory");
#endif
#endif
}

// ---------------------------------------------------------------------------
// WMMA helpers (CDNA5 16x16x32 bf16, fp32 accumulate)
// ---------------------------------------------------------------------------
__device__ __forceinline__ v8f wmma_bf16(v16bf a, v16bf b, v8f c) {
  // (neg_a, A, neg_b, B, c_mod, C, reuse_a, reuse_b)
  return __builtin_amdgcn_wmma_f32_16x16x32_bf16(false, a, false, b,
                                                 (short)0, c, false, false);
}

// A fragment: 16x32 bf16 tile from row-major matrix (ld = row stride, elems).
// ISA layout: lane m = lane&15 holds row m; elements 0..7 hold K = kh+0..7,
// elements 8..15 hold K = 16+kh+0..7, where kh = (lane>>4)*8.
__device__ __forceinline__ v16bf load_a_frag(const bf16_t* p, int ld,
                                             int row, int k0, int lane) {
  int m  = lane & 15;
  int kh = (lane >> 4) << 3;   // 0 or 8
  const bf16_t* r = p + (size_t)(row + m) * ld + k0 + kh;
  v16bf f;
  *(uint4*)&f         = *(const uint4*)(r);        // K = kh .. kh+7
  *(((uint4*)&f) + 1) = *(const uint4*)(r + 16);   // K = 16+kh .. 16+kh+7
  return f;
}

// B fragment: 32x16 bf16 tile, sourced from B^T stored row-major as [N x K]
// (ld = K stride). ISA layout: lane holds column n = lane&15; elements 0..7
// hold K = kh+0..7, elements 8..15 hold K = 16+kh+0..7 (kh = (lane>>4)*8).
__device__ __forceinline__ v16bf load_bT_frag(const bf16_t* pT, int ld,
                                              int col, int k0, int lane) {
  int n  = lane & 15;
  int kh = (lane >> 4) << 3;   // 0 or 8
  const bf16_t* r = pT + (size_t)(col + n) * ld + k0 + kh;
  v16bf f;
  *(uint4*)&f         = *(const uint4*)(r);
  *(((uint4*)&f) + 1) = *(const uint4*)(r + 16);
  return f;
}

// ---------------------------------------------------------------------------
// Kernel 1a: elementwise f32 -> bf16
// ---------------------------------------------------------------------------
__global__ void cvt_f32_bf16(const float* __restrict__ in,
                             bf16_t* __restrict__ out, int n) {
  int i = blockIdx.x * blockDim.x + threadIdx.x;
  if (i < n) out[i] = (bf16_t)in[i];
}

// ---------------------------------------------------------------------------
// Kernel 1b: per-head weight transpose+convert: W[h][e][d] -> Wt[h][d][e]
// (Wo handled with nh=1, ne=HD, nd=E: Wo[k][e] -> Wot[e][k])
// ---------------------------------------------------------------------------
__global__ void transpose_w(const float* __restrict__ w,
                            bf16_t* __restrict__ wt,
                            int nh, int ne, int nd) {
  int i = blockIdx.x * blockDim.x + threadIdx.x;
  int total = nh * ne * nd;
  if (i >= total) return;
  int h   = i / (ne * nd);
  int rem = i - h * (ne * nd);
  int e   = rem / nd;
  int d   = rem - e * nd;
  wt[(size_t)h * nd * ne + (size_t)d * ne + e] = (bf16_t)w[i];
}

// ---------------------------------------------------------------------------
// Kernel 2: QKV projection GEMM. 4 waves/block; 32x128 C tile per block.
// The shared 32x32 A tile is async-staged into double-buffered LDS; each
// wave owns a 32-column strip of C (2x2 WMMA tiles).
//   C[s, h*64+d] = sum_e xb[s,e] * Wt[h][d][e]  + bias[h][d]
// transpose_out==0: write [B,H,S,Dh] (Q, K);  ==1: write [B,H,Dh,S] (V^T).
// ---------------------------------------------------------------------------
__global__ void qkv_gemm(const bf16_t* __restrict__ xb,   // [BS, E]
                         const bf16_t* __restrict__ wt,   // [H, Dh, E]
                         const float*  __restrict__ bias, // [H, Dh]
                         bf16_t* __restrict__ out,
                         int transpose_out) {
  __shared__ __align__(16) bf16_t Abuf[2][32 * 32];  // 2 x 2KB

  int tid  = threadIdx.x;
  int lane = tid & 31;
  int wave = tid >> 5;                  // 0..3
  int row0 = blockIdx.x * 32;           // global row index into [BS]
  int col0 = blockIdx.y * 128 + wave * 32;  // global col index into [HD]
  int h  = col0 / Dh_;
  int d0 = col0 % Dh_;
  const bf16_t* bt = wt + (size_t)h * Dh_ * E_ + (size_t)d0 * E_;

  // Staging assignment: 128 threads x one b128 = 32 rows x 32 K of bf16.
  int srow   = tid >> 2;
  int schunk = (tid & 3) * 8;
  const bf16_t* sbase = xb + (size_t)(row0 + srow) * E_ + schunk;

  // Prologue: stage k-step 0.
  async_copy_b128(sbase, &Abuf[0][srow * 32 + schunk]);
  wait_async();
  __syncthreads();

  v8f acc[2][2] = {};
  constexpr int NSTEP = E_ / 32;
  for (int it = 0; it < NSTEP; ++it) {
    int cur = it & 1;
    if (it + 1 < NSTEP)   // overlap: load(it+1) while computing(it)
      async_copy_b128(sbase + (it + 1) * 32,
                      &Abuf[cur ^ 1][srow * 32 + schunk]);
    int k0 = it * 32;
    if (k0 + 32 < E_)
      __builtin_prefetch(bt + (size_t)(lane & 15) * E_ + k0 + 32, 0, 1);

    v16bf a0 = load_a_frag(&Abuf[cur][0], 32, 0,  0, lane);  // ds_load_b128
    v16bf a1 = load_a_frag(&Abuf[cur][0], 32, 16, 0, lane);
    v16bf b0 = load_bT_frag(bt, E_, 0,  k0, lane);
    v16bf b1 = load_bT_frag(bt, E_, 16, k0, lane);
    acc[0][0] = wmma_bf16(a0, b0, acc[0][0]);
    acc[0][1] = wmma_bf16(a0, b1, acc[0][1]);
    acc[1][0] = wmma_bf16(a1, b0, acc[1][0]);
    acc[1][1] = wmma_bf16(a1, b1, acc[1][1]);

    wait_async();
    __syncthreads();
  }

  // Epilogue: C layout -> lane holds (row mh+r, col n) per v8f element r.
  int n  = lane & 15;
  int mh = (lane >> 4) * 8;
  int b      = row0 / S_;
  int s_base = row0 % S_;
  size_t head_base = ((size_t)b * H_ + h) * (size_t)S_ * Dh_;
  #pragma unroll
  for (int im = 0; im < 2; ++im)
    #pragma unroll
    for (int jn = 0; jn < 2; ++jn)
      #pragma unroll
      for (int r = 0; r < 8; ++r) {
        int s = s_base + im * 16 + mh + r;
        int d = d0 + jn * 16 + n;
        float v = acc[im][jn][r] + bias[h * Dh_ + d];
        if (transpose_out)
          out[head_base + (size_t)d * S_ + s] = (bf16_t)v;  // V^T [Dh,S]
        else
          out[head_base + (size_t)s * Dh_ + d] = (bf16_t)v; // Q/K [S,Dh]
      }
}

// ---------------------------------------------------------------------------
// Kernel 3: flash-style attention. 4 waves/block handle 64 query rows of one
// (b,h); K (32x64) and V^T (64x32) chunk tiles are async-staged into
// double-buffered LDS and shared by all 4 waves; online softmax per wave.
// ---------------------------------------------------------------------------
__global__ void attn_kernel(const bf16_t* __restrict__ Qb, // [B,H,S,Dh]
                            const bf16_t* __restrict__ Kb, // [B,H,S,Dh]
                            const bf16_t* __restrict__ Vt, // [B,H,Dh,S]
                            bf16_t* __restrict__ Zc) {     // [B,S,HD]
  __shared__ __align__(16) bf16_t Kbuf[2][32 * 64];  // 2 x 4KB
  __shared__ __align__(16) bf16_t Vbuf[2][64 * 32];  // 2 x 4KB
  __shared__ __align__(16) bf16_t Pbuf[4][16 * 32];  // per-wave P staging

  int tid  = threadIdx.x;
  int lane = tid & 31;
  int wave = tid >> 5;
  int s0 = blockIdx.x * 64 + wave * 16;
  int h  = blockIdx.y;
  int b  = blockIdx.z;
  size_t hb = ((size_t)b * H_ + h) * (size_t)S_ * Dh_;
  const bf16_t* Qh = Qb + hb;
  const bf16_t* Kh = Kb + hb;
  const bf16_t* Vh = Vt + hb;

  // Stage one 32-key chunk: K rows [t0..t0+31] x 64d, V^T rows [0..63]d x 32t.
  auto stage = [&](int buf, int t0) {
    #pragma unroll
    for (int i = 0; i < 2; ++i) {     // 256 b128 chunks of K / 128 threads
      int idx = tid + i * 128;
      int tr = idx >> 3, ch = (idx & 7) * 8;
      async_copy_b128(Kh + (size_t)(t0 + tr) * Dh_ + ch,
                      &Kbuf[buf][tr * 64 + ch]);
    }
    #pragma unroll
    for (int i = 0; i < 2; ++i) {     // 256 b128 chunks of V / 128 threads
      int idx = tid + i * 128;
      int dr = idx >> 2, ch = (idx & 3) * 8;
      async_copy_b128(Vh + (size_t)dr * S_ + t0 + ch,
                      &Vbuf[buf][dr * 32 + ch]);
    }
  };

  // Q tile (16 x 64) as two A fragments, loaded once per wave.
  v16bf aq0 = load_a_frag(Qh, Dh_, s0, 0,  lane);
  v16bf aq1 = load_a_frag(Qh, Dh_, s0, 32, lane);

  v8f o[4] = {};
  float m_i[8], l_i[8];
  #pragma unroll
  for (int r = 0; r < 8; ++r) { m_i[r] = -1e30f; l_i[r] = 0.0f; }

  const float scale = 0.125f;  // 1/sqrt(Dh)
  int n  = lane & 15;
  int mh = (lane >> 4) * 8;

  stage(0, 0);
  wait_async();
  __syncthreads();

  constexpr int NSTEP = S_ / 32;      // 64 key chunks
  for (int it = 0; it < NSTEP; ++it) {
    int cur = it & 1;
    if (it + 1 < NSTEP)
      stage(cur ^ 1, (it + 1) * 32);  // overlap with compute below

    const bf16_t* Kt = &Kbuf[cur][0];  // [32 t][64 d]
    const bf16_t* Vv = &Vbuf[cur][0];  // [64 d][32 t]

    // Scores tile: 16 x 32 = two 16x16 WMMA C tiles, K-dim = Dh = 64.
    v8f sc0 = {}, sc1 = {};
    sc0 = wmma_bf16(aq0, load_bT_frag(Kt, 64, 0,  0,  lane), sc0);
    sc0 = wmma_bf16(aq1, load_bT_frag(Kt, 64, 0,  32, lane), sc0);
    sc1 = wmma_bf16(aq0, load_bT_frag(Kt, 64, 16, 0,  lane), sc1);
    sc1 = wmma_bf16(aq1, load_bT_frag(Kt, 64, 16, 32, lane), sc1);

    // Online softmax update (per-row stats live in the 8 C-row slots).
    #pragma unroll
    for (int r = 0; r < 8; ++r) {
      float a0s = sc0[r] * scale;
      float a1s = sc1[r] * scale;
      float rm = fmaxf(a0s, a1s);
      rm = fmaxf(rm, __shfl_xor(rm, 1, 32));
      rm = fmaxf(rm, __shfl_xor(rm, 2, 32));
      rm = fmaxf(rm, __shfl_xor(rm, 4, 32));
      rm = fmaxf(rm, __shfl_xor(rm, 8, 32));
      float m_new = fmaxf(m_i[r], rm);
      float alpha = __expf(m_i[r] - m_new);
      float p0 = __expf(a0s - m_new);
      float p1 = __expf(a1s - m_new);
      float rs = p0 + p1;
      rs += __shfl_xor(rs, 1, 32);
      rs += __shfl_xor(rs, 2, 32);
      rs += __shfl_xor(rs, 4, 32);
      rs += __shfl_xor(rs, 8, 32);
      l_i[r] = l_i[r] * alpha + rs;
      m_i[r] = m_new;
      // Stage P (bf16) into this wave's LDS slab for the A-layout reload.
      Pbuf[wave][(mh + r) * 32 + n]      = (bf16_t)p0;
      Pbuf[wave][(mh + r) * 32 + 16 + n] = (bf16_t)p1;
      #pragma unroll
      for (int j = 0; j < 4; ++j) o[j][r] *= alpha;
    }

    // Reload P (16x32) as a WMMA A fragment (same-wave LDS ops are in-order).
    v16bf ap;
    {
      int m  = lane & 15;
      int kh = (lane >> 4) << 3;
      const bf16_t* pr = &Pbuf[wave][m * 32 + kh];
      *(uint4*)&ap         = *(const uint4*)(pr);
      *(((uint4*)&ap) + 1) = *(const uint4*)(pr + 16);
    }

    // O += P @ V : K-dim = 32 keys, 4 output d-subtiles of 16.
    #pragma unroll
    for (int j = 0; j < 4; ++j)
      o[j] = wmma_bf16(ap, load_bT_frag(Vv, 32, j * 16, 0, lane), o[j]);

    wait_async();
    __syncthreads();
  }

  // Normalize by row sums and store heads concatenated: Zc[b][s][h*64+d].
  #pragma unroll
  for (int j = 0; j < 4; ++j)
    #pragma unroll
    for (int r = 0; r < 8; ++r) {
      float v = o[j][r] / l_i[r];
      int s = s0 + mh + r;
      int d = j * 16 + n;
      Zc[((size_t)b * S_ + s) * HD_ + h * Dh_ + d] = (bf16_t)v;
    }
}

// ---------------------------------------------------------------------------
// Kernel 4: output projection.  out[s, e] = Zc[s, :] @ Wo[:, e] + bo[e]
// Same async-LDS cooperative structure as qkv_gemm; fp32 output + bias.
// ---------------------------------------------------------------------------
__global__ void out_gemm(const bf16_t* __restrict__ zc,   // [BS, HD]
                         const bf16_t* __restrict__ wot,  // [E, HD]
                         const float*  __restrict__ bo,   // [E]
                         float* __restrict__ out) {       // [BS, E]
  __shared__ __align__(16) bf16_t Abuf[2][32 * 32];

  int tid  = threadIdx.x;
  int lane = tid & 31;
  int wave = tid >> 5;
  int row0 = blockIdx.x * 32;
  int col0 = blockIdx.y * 128 + wave * 32;

  int srow   = tid >> 2;
  int schunk = (tid & 3) * 8;
  const bf16_t* sbase = zc + (size_t)(row0 + srow) * HD_ + schunk;

  async_copy_b128(sbase, &Abuf[0][srow * 32 + schunk]);
  wait_async();
  __syncthreads();

  v8f acc[2][2] = {};
  constexpr int NSTEP = HD_ / 32;
  for (int it = 0; it < NSTEP; ++it) {
    int cur = it & 1;
    if (it + 1 < NSTEP)
      async_copy_b128(sbase + (it + 1) * 32,
                      &Abuf[cur ^ 1][srow * 32 + schunk]);
    int k0 = it * 32;
    if (k0 + 32 < HD_)
      __builtin_prefetch(wot + (size_t)(col0 + (lane & 15)) * HD_ + k0 + 32, 0, 1);

    v16bf a0 = load_a_frag(&Abuf[cur][0], 32, 0,  0, lane);
    v16bf a1 = load_a_frag(&Abuf[cur][0], 32, 16, 0, lane);
    v16bf b0 = load_bT_frag(wot, HD_, col0,      k0, lane);
    v16bf b1 = load_bT_frag(wot, HD_, col0 + 16, k0, lane);
    acc[0][0] = wmma_bf16(a0, b0, acc[0][0]);
    acc[0][1] = wmma_bf16(a0, b1, acc[0][1]);
    acc[1][0] = wmma_bf16(a1, b0, acc[1][0]);
    acc[1][1] = wmma_bf16(a1, b1, acc[1][1]);

    wait_async();
    __syncthreads();
  }

  int n  = lane & 15;
  int mh = (lane >> 4) * 8;
  #pragma unroll
  for (int im = 0; im < 2; ++im)
    #pragma unroll
    for (int jn = 0; jn < 2; ++jn)
      #pragma unroll
      for (int r = 0; r < 8; ++r) {
        int row = row0 + im * 16 + mh + r;
        int col = col0 + jn * 16 + n;
        out[(size_t)row * E_ + col] = acc[im][jn][r] + bo[col];
      }
}

// ---------------------------------------------------------------------------
// Host-side launch
// ---------------------------------------------------------------------------
extern "C" void kernel_launch(void* const* d_in, const int* in_sizes, int n_in,
                              void* d_out, int out_size, void* d_ws, size_t ws_size,
                              hipStream_t stream) {
  (void)in_sizes; (void)n_in; (void)out_size; (void)ws_size;

  const float* x  = (const float*)d_in[0];
  const float* Wq = (const float*)d_in[1];
  const float* bq = (const float*)d_in[2];
  const float* Wk = (const float*)d_in[3];
  const float* bk = (const float*)d_in[4];
  const float* Wv = (const float*)d_in[5];
  const float* bv = (const float*)d_in[6];
  const float* Wo = (const float*)d_in[7];
  const float* bo = (const float*)d_in[8];
  float* out = (float*)d_out;

  // Workspace carve-up (256B aligned chunks), total ~92 MB.
  char* w = (char*)d_ws;
  auto take = [&](size_t bytes) -> void* {
    void* p = (void*)w;
    w += (bytes + 255) & ~(size_t)255;
    return p;
  };
  bf16_t* xb  = (bf16_t*)take((size_t)BS_ * E_ * sizeof(bf16_t));
  bf16_t* Wqt = (bf16_t*)take((size_t)H_ * Dh_ * E_ * sizeof(bf16_t));
  bf16_t* Wkt = (bf16_t*)take((size_t)H_ * Dh_ * E_ * sizeof(bf16_t));
  bf16_t* Wvt = (bf16_t*)take((size_t)H_ * Dh_ * E_ * sizeof(bf16_t));
  bf16_t* Wot = (bf16_t*)take((size_t)E_ * HD_ * sizeof(bf16_t));
  bf16_t* Qb  = (bf16_t*)take((size_t)B_ * H_ * S_ * Dh_ * sizeof(bf16_t));
  bf16_t* Kb  = (bf16_t*)take((size_t)B_ * H_ * S_ * Dh_ * sizeof(bf16_t));
  bf16_t* Vt  = (bf16_t*)take((size_t)B_ * H_ * S_ * Dh_ * sizeof(bf16_t));
  bf16_t* Zc  = (bf16_t*)take((size_t)B_ * S_ * HD_ * sizeof(bf16_t));

  // 1) Conversions / weight transposes.
  {
    int n = BS_ * E_;
    cvt_f32_bf16<<<(n + 255) / 256, 256, 0, stream>>>(x, xb, n);
    int nw = H_ * E_ * Dh_;
    transpose_w<<<(nw + 255) / 256, 256, 0, stream>>>(Wq, Wqt, H_, E_, Dh_);
    transpose_w<<<(nw + 255) / 256, 256, 0, stream>>>(Wk, Wkt, H_, E_, Dh_);
    transpose_w<<<(nw + 255) / 256, 256, 0, stream>>>(Wv, Wvt, H_, E_, Dh_);
    int no = HD_ * E_;
    transpose_w<<<(no + 255) / 256, 256, 0, stream>>>(Wo, Wot, 1, HD_, E_);
  }

  // 2) QKV projections (WMMA bf16 + async LDS A-tiles). V pre-transposed.
  {
    dim3 grid(BS_ / 32, HD_ / 128);
    qkv_gemm<<<grid, 128, 0, stream>>>(xb, Wqt, bq, Qb, 0);
    qkv_gemm<<<grid, 128, 0, stream>>>(xb, Wkt, bk, Kb, 0);
    qkv_gemm<<<grid, 128, 0, stream>>>(xb, Wvt, bv, Vt, 1);
  }

  // 3) Flash attention (WMMA bf16 + online softmax + async K/V tiles).
  {
    dim3 grid(S_ / 64, H_, B_);
    attn_kernel<<<grid, 128, 0, stream>>>(Qb, Kb, Vt, Zc);
  }

  // 4) Output projection (WMMA bf16 -> fp32 + bias).
  {
    dim3 grid(BS_ / 32, E_ / 128);
    out_gemm<<<grid, 128, 0, stream>>>(Zc, Wot, bo, out);
  }
}
